// MoeFcTokensParallel_41979010351184
// MI455X (gfx1250) — compile-verified, matrix operand order checked by reading, not run on hardware
//
#include <hip/hip_runtime.h>
#include <hip/hip_bf16.h>

// Problem constants (match reference)
constexpr int BB = 4;     // batch
constexpr int SS = 2048;  // tokens
constexpr int DD = 1024;  // model dim
constexpr int OO = 1024;  // out dim
constexpr int EE = 16;    // experts
constexpr int KK = 512;   // top-k tokens per expert

typedef __attribute__((ext_vector_type(16))) __bf16 v16bf;
typedef __attribute__((ext_vector_type(2)))  __bf16 v2bf;
typedef __attribute__((ext_vector_type(8)))  float  v8f;
typedef __attribute__((ext_vector_type(4)))  float  v4f;
typedef __attribute__((ext_vector_type(4)))  unsigned int v4u;
typedef __attribute__((ext_vector_type(2)))  unsigned int v2u;

__device__ __forceinline__ unsigned int pk2(float a, float b) {
#if __has_builtin(__builtin_amdgcn_cvt_pk_bf16_f32)
    v2bf p = __builtin_amdgcn_cvt_pk_bf16_f32(a, b);
    return __builtin_bit_cast(unsigned int, p);
#else
    unsigned int ua = __float_as_uint(a);
    ua += 0x7FFFu + ((ua >> 16) & 1u);
    unsigned int ub = __float_as_uint(b);
    ub += 0x7FFFu + ((ub >> 16) & 1u);
    return (ua >> 16) | (ub & 0xFFFF0000u);
#endif
}

// ---------------------------------------------------------------------------
// Stage 1: gate logits, written transposed [B,E,S]; one block per token
// ---------------------------------------------------------------------------
__global__ __launch_bounds__(128) void gate_logits(const float* __restrict__ x,
                                                   const float* __restrict__ gw,
                                                   const float* __restrict__ gb,
                                                   float* __restrict__ logitsT) {
    const int token = blockIdx.x;            // 0 .. B*S-1
    const int b = token / SS, s = token % SS;
    __shared__ float xs[DD];
    __shared__ float red[EE][9];
    const int t = threadIdx.x;

    const float* xr = x + (size_t)token * DD;
    for (int i = t; i < DD / 4; i += 128)
        ((v4f*)xs)[i] = ((const v4f*)xr)[i];
    __syncthreads();

    const int e = t & 15, c = t >> 4;
    float p = 0.f;
    const int d0 = c * 128;
    for (int d = 0; d < 128; ++d)
        p += xs[d0 + d] * gw[(size_t)(d0 + d) * EE + e];
    red[e][c] = p;
    __syncthreads();
    if (t < EE) {
        float sum = gb[t];
        for (int cc = 0; cc < 8; ++cc) sum += red[t][cc];
        logitsT[((size_t)b * EE + t) * SS + s] = sum;
    }
}

// ---------------------------------------------------------------------------
// Stage 2: softmax over TOKEN axis, one block per (b,e)
// ---------------------------------------------------------------------------
__global__ __launch_bounds__(256) void softmax_tok(const float* __restrict__ logitsT,
                                                   float* __restrict__ probsT) {
    const int be = blockIdx.x;
    const float* L = logitsT + (size_t)be * SS;
    float* P = probsT + (size_t)be * SS;
    __shared__ float red[256];
    const int t = threadIdx.x;

    float mx = -3.4e38f;
    for (int s = t; s < SS; s += 256) mx = fmaxf(mx, L[s]);
    red[t] = mx; __syncthreads();
    for (int off = 128; off > 0; off >>= 1) {
        if (t < off) red[t] = fmaxf(red[t], red[t + off]);
        __syncthreads();
    }
    mx = red[0]; __syncthreads();

    float sm = 0.f;
    for (int s = t; s < SS; s += 256) sm += __expf(L[s] - mx);
    red[t] = sm; __syncthreads();
    for (int off = 128; off > 0; off >>= 1) {
        if (t < off) red[t] += red[t + off];
        __syncthreads();
    }
    const float inv = 1.f / red[0];
    for (int s = t; s < SS; s += 256) P[s] = __expf(L[s] - mx) * inv;
}

// ---------------------------------------------------------------------------
// Stage 3: exact top-K per (b,e) via bitwise threshold search on uint keys
// ---------------------------------------------------------------------------
__global__ __launch_bounds__(256) void topk_sel(const float* __restrict__ probsT,
                                                int* __restrict__ idxW,
                                                float* __restrict__ recipW) {
    const int be = blockIdx.x;
    const float* P = probsT + (size_t)be * SS;
    __shared__ float pv[SS];
    __shared__ int red[256];
    __shared__ int cnt;
    const int t = threadIdx.x;

    for (int s = t; s < SS; s += 256) pv[s] = P[s];
    __syncthreads();

    unsigned int T = 0;
    for (int bit = 31; bit >= 0; --bit) {
        const unsigned int cand = T | (1u << bit);
        int c = 0;
        for (int s = t; s < SS; s += 256)
            c += (__float_as_uint(pv[s]) >= cand) ? 1 : 0;
        red[t] = c; __syncthreads();
        for (int off = 128; off > 0; off >>= 1) {
            if (t < off) red[t] += red[t + off];
            __syncthreads();
        }
        const int total = red[0];
        __syncthreads();
        if (total >= KK) T = cand;
    }

    if (t == 0) cnt = 0;
    __syncthreads();
    int* myIdx = idxW + (size_t)be * KK;
    float* myRec = recipW + (size_t)be * KK;
    for (int s = t; s < SS; s += 256) {
        if (__float_as_uint(pv[s]) > T) {
            int p = atomicAdd(&cnt, 1);
            myIdx[p] = s; myRec[p] = 1.f / pv[s];
        }
    }
    __syncthreads();
    for (int s = t; s < SS; s += 256) {
        if (__float_as_uint(pv[s]) == T) {
            int p = atomicAdd(&cnt, 1);
            if (p < KK) { myIdx[p] = s; myRec[p] = 1.f / pv[s]; }
        }
    }
}

// ---------------------------------------------------------------------------
// One-time conversion passes (bandwidth-bound):
//   cvt_x : x f32 [B,S,D]    -> bf16 [B,S,D]
//   cvt_wT: w f32 [E,D,O]    -> bf16 [E,O,D]  (transposed for column loads)
// ---------------------------------------------------------------------------
__global__ __launch_bounds__(256) void cvt_x(const float* __restrict__ x,
                                             unsigned short* __restrict__ xb) {
    const size_t i = (size_t)blockIdx.x * 256 + threadIdx.x;   // 4 floats each
    v4f f = ((const v4f*)x)[i];
    v2u p; p.x = pk2(f.x, f.y); p.y = pk2(f.z, f.w);
    ((v2u*)xb)[i] = p;
}

__global__ __launch_bounds__(256) void cvt_wT(const float* __restrict__ w,
                                              unsigned short* __restrict__ wT) {
    // one thread produces 8 contiguous bf16 along d for one output column o
    const size_t gi = (size_t)blockIdx.x * 256 + threadIdx.x;
    const int e  = (int)(gi >> 17);            // O*D/8 = 131072 chunks/expert
    const int rem = (int)(gi & 131071);
    const int o  = rem & 1023;                 // fast index -> coalesced reads
    const int d0 = (rem >> 10) * 8;
    const float* src = w + ((size_t)e * DD + d0) * OO + o;
    v4u p;
    p.x = pk2(src[0],            src[(size_t)OO]);
    p.y = pk2(src[2*(size_t)OO], src[3*(size_t)OO]);
    p.z = pk2(src[4*(size_t)OO], src[5*(size_t)OO]);
    p.w = pk2(src[6*(size_t)OO], src[7*(size_t)OO]);
    *(v4u*)&wT[((size_t)e * OO + o) * DD + d0] = p;
}

// ---------------------------------------------------------------------------
// Stage 4 (fast path): gathered bf16 GEMM per (b,e): [512 x 1024]@[1024 x 1024]
// 128x256 block tile, 8 wave32s in 2x4, wave tile 64x64 = 4x4 WMMA 16x16x32.
// Pointer-swapped double buffer, unroll(1) to keep VGPRs < 256.
// ---------------------------------------------------------------------------
union FragBF { v16bf v; v4u u[2]; };

constexpr int LST = 40;         // LDS row stride in bf16 elems (80B, 16B-aligned)
constexpr int BM = 128, BN = 256, BK = 32;
constexpr int NKT = DD / BK;    // 32 k-steps

__global__ __launch_bounds__(256) void moe_gemm_bf(const unsigned short* __restrict__ xb,
                                                   const unsigned short* __restrict__ wT,
                                                   const float* __restrict__ bias,
                                                   const int* __restrict__ idxW,
                                                   const float* __restrict__ recipW,
                                                   float* __restrict__ out) {
    const int blk = blockIdx.x;
    const int be   = blk >> 4;
    const int tile = blk & 15;
    const int mt = tile >> 2;          // M tiles of 128 over K=512
    const int nt = tile & 3;           // N tiles of 256 over O=1024
    const int b  = be >> 4, e = be & 15;

    __shared__ unsigned short Al[2][BM * LST];
    __shared__ unsigned short Bl[2][BN * LST];
    __shared__ int   sTok[BM];
    __shared__ float sRec[BM];

    const int tid = threadIdx.x;
    if (tid < BM) {
        const size_t base = (size_t)be * KK + (size_t)mt * BM + tid;
        sTok[tid] = idxW[base];
        sRec[tid] = recipW[base];
    }
    __syncthreads();

    // A: 2 threads per gathered row, 16 bf16 (32B) each per k-step
    const int ar = tid >> 1;
    const int ah = (tid & 1) * 16;
    const unsigned short* xrow = xb + ((size_t)b * SS + sTok[ar]) * DD + ah;
    // B: thread owns column o = nt*256+tid, 32 bf16 (64B contiguous) per k-step
    const unsigned short* wcol = wT + ((size_t)e * OO + (size_t)nt * BN + tid) * DD;

    const int wid = tid >> 5, lane = tid & 31;
    const int wm = wid & 1, wn = wid >> 1;     // 2 waves on M, 4 on N
    const int l15 = lane & 15, hi = lane >> 4;

    v8f acc[4][4];
    #pragma unroll
    for (int mi = 0; mi < 4; ++mi)
        #pragma unroll
        for (int ni = 0; ni < 4; ++ni)
            acc[mi][ni] = (v8f){0.f,0.f,0.f,0.f,0.f,0.f,0.f,0.f};

    unsigned short* Acur = Al[0]; unsigned short* Anxt = Al[1];
    unsigned short* Bcur = Bl[0]; unsigned short* Bnxt = Bl[1];

    v4u ra[2], rb[4];
    // prologue: fetch + stage k-step 0 into current buffers
    ra[0] = *(const v4u*)(xrow);
    ra[1] = *(const v4u*)(xrow + 8);
    #pragma unroll
    for (int g = 0; g < 4; ++g) rb[g] = *(const v4u*)(wcol + g * 8);
    *(v4u*)&Acur[ar * LST + ah]     = ra[0];
    *(v4u*)&Acur[ar * LST + ah + 8] = ra[1];
    #pragma unroll
    for (int g = 0; g < 4; ++g) *(v4u*)&Bcur[tid * LST + g * 8] = rb[g];

    #pragma unroll 1
    for (int kt = 0; kt < NKT; ++kt) {
        __syncthreads();
        const bool more = (kt + 1) < NKT;

        if (more) {                      // next tile's global loads, issued early
            const int d1 = (kt + 1) * BK;
            ra[0] = *(const v4u*)(xrow + d1);
            ra[1] = *(const v4u*)(xrow + d1 + 8);
            #pragma unroll
            for (int g = 0; g < 4; ++g) rb[g] = *(const v4u*)(wcol + d1 + g * 8);
        }

        // hold all 4 B fragments; stream A fragments one at a time
        FragBF fb[4];
        #pragma unroll
        for (int ni = 0; ni < 4; ++ni) {
            const int c = wn * 64 + ni * 16 + l15;
            const v4u* p = (const v4u*)&Bcur[c * LST + hi * 16];
            fb[ni].u[0] = p[0]; fb[ni].u[1] = p[1];
        }
        #pragma unroll
        for (int mi = 0; mi < 4; ++mi) {
            FragBF fa;
            const int r = wm * 64 + mi * 16 + l15;
            const v4u* p = (const v4u*)&Acur[r * LST + hi * 16];
            fa.u[0] = p[0]; fa.u[1] = p[1];
            #pragma unroll
            for (int ni = 0; ni < 4; ++ni)
                acc[mi][ni] = __builtin_amdgcn_wmma_f32_16x16x32_bf16(
                    false, fa.v, false, fb[ni].v, (short)0, acc[mi][ni],
                    false, false);
        }

        if (more) {                      // stage next tile into the other buffer
            *(v4u*)&Anxt[ar * LST + ah]     = ra[0];
            *(v4u*)&Anxt[ar * LST + ah + 8] = ra[1];
            #pragma unroll
            for (int g = 0; g < 4; ++g) *(v4u*)&Bnxt[tid * LST + g * 8] = rb[g];
        }
        unsigned short* tp;
        tp = Acur; Acur = Anxt; Anxt = tp;
        tp = Bcur; Bcur = Bnxt; Bnxt = tp;
    }

    const float bs = bias[e];
    float* outB = out + (size_t)b * SS * OO;
    #pragma unroll
    for (int mi = 0; mi < 4; ++mi) {
        #pragma unroll
        for (int v = 0; v < 8; ++v) {
            const int rl = wm * 64 + mi * 16 + v + 8 * hi;
            const int tok = sTok[rl];
            const float rc = sRec[rl];
            #pragma unroll
            for (int ni = 0; ni < 4; ++ni) {
                const int gc = nt * BN + wn * 64 + ni * 16 + l15;
                const float val = (acc[mi][ni][v] + bs) * rc;
                atomicAdd(outB + (size_t)tok * OO + gc, val);
            }
        }
    }
}

// ---------------------------------------------------------------------------
// Stage 4 (fallback, small ws): f32-staging GEMM with in-loop conversion
// ---------------------------------------------------------------------------
__global__ __launch_bounds__(256) void moe_gemm(const float* __restrict__ x,
                                                const float* __restrict__ w,
                                                const float* __restrict__ bias,
                                                const int* __restrict__ idxW,
                                                const float* __restrict__ recipW,
                                                float* __restrict__ out) {
    const int blk = blockIdx.x;
    const int be   = blk >> 4;
    const int tile = blk & 15;
    const int mt = tile >> 2;
    const int nt = tile & 3;
    const int b  = be >> 4, e = be & 15;

    __shared__ unsigned short Al[2][BM * LST];
    __shared__ unsigned short Bl[2][BN * LST];
    __shared__ int   sTok[BM];
    __shared__ float sRec[BM];

    const int tid = threadIdx.x;
    if (tid < BM) {
        const size_t base = (size_t)be * KK + (size_t)mt * BM + tid;
        sTok[tid] = idxW[base];
        sRec[tid] = recipW[base];
    }
    __syncthreads();

    const int ar = tid >> 1;
    const int ah = (tid & 1) * 16;
    const float* xrow = x + ((size_t)b * SS + sTok[ar]) * DD;
    const float* wcol = w + (size_t)e * DD * OO + (size_t)nt * BN + tid;

    const int wid = tid >> 5, lane = tid & 31;
    const int wm = wid & 1, wn = wid >> 1;
    const int l15 = lane & 15, hi = lane >> 4;

    v8f acc[4][4];
    #pragma unroll
    for (int mi = 0; mi < 4; ++mi)
        #pragma unroll
        for (int ni = 0; ni < 4; ++ni)
            acc[mi][ni] = (v8f){0.f,0.f,0.f,0.f,0.f,0.f,0.f,0.f};

    unsigned short* Acur = Al[0]; unsigned short* Anxt = Al[1];
    unsigned short* Bcur = Bl[0]; unsigned short* Bnxt = Bl[1];

    v2u ra[4], rb[8];
    #pragma unroll
    for (int j = 0; j < 4; ++j) {
        v4f f = *(const v4f*)(xrow + ah + j * 4);
        ra[j].x = pk2(f.x, f.y); ra[j].y = pk2(f.z, f.w);
    }
    #pragma unroll
    for (int g = 0; g < 8; ++g) {
        const size_t dbase = (size_t)(g * 4) * OO;
        rb[g].x = pk2(wcol[dbase], wcol[dbase + OO]);
        rb[g].y = pk2(wcol[dbase + 2 * OO], wcol[dbase + 3 * OO]);
    }
    #pragma unroll
    for (int j = 0; j < 4; ++j)
        *(v2u*)&Acur[ar * LST + ah + j * 4] = ra[j];
    #pragma unroll
    for (int g = 0; g < 8; ++g)
        *(v2u*)&Bcur[tid * LST + g * 4] = rb[g];

    #pragma unroll 1
    for (int kt = 0; kt < NKT; ++kt) {
        __syncthreads();
        const bool more = (kt + 1) < NKT;

        if (more) {
            const int d1 = (kt + 1) * BK;
            #pragma unroll
            for (int j = 0; j < 4; ++j) {
                v4f f = *(const v4f*)(xrow + d1 + ah + j * 4);
                ra[j].x = pk2(f.x, f.y); ra[j].y = pk2(f.z, f.w);
            }
            #pragma unroll
            for (int g = 0; g < 8; ++g) {
                const size_t dbase = (size_t)(d1 + g * 4) * OO;
                rb[g].x = pk2(wcol[dbase], wcol[dbase + OO]);
                rb[g].y = pk2(wcol[dbase + 2 * OO], wcol[dbase + 3 * OO]);
            }
        }

        FragBF fb[4];
        #pragma unroll
        for (int ni = 0; ni < 4; ++ni) {
            const int c = wn * 64 + ni * 16 + l15;
            const v4u* p = (const v4u*)&Bcur[c * LST + hi * 16];
            fb[ni].u[0] = p[0]; fb[ni].u[1] = p[1];
        }
        #pragma unroll
        for (int mi = 0; mi < 4; ++mi) {
            FragBF fa;
            const int r = wm * 64 + mi * 16 + l15;
            const v4u* p = (const v4u*)&Acur[r * LST + hi * 16];
            fa.u[0] = p[0]; fa.u[1] = p[1];
            #pragma unroll
            for (int ni = 0; ni < 4; ++ni)
                acc[mi][ni] = __builtin_amdgcn_wmma_f32_16x16x32_bf16(
                    false, fa.v, false, fb[ni].v, (short)0, acc[mi][ni],
                    false, false);
        }

        if (more) {
            #pragma unroll
            for (int j = 0; j < 4; ++j)
                *(v2u*)&Anxt[ar * LST + ah + j * 4] = ra[j];
            #pragma unroll
            for (int g = 0; g < 8; ++g)
                *(v2u*)&Bnxt[tid * LST + g * 4] = rb[g];
        }
        unsigned short* tp;
        tp = Acur; Acur = Anxt; Anxt = tp;
        tp = Bcur; Bcur = Bnxt; Bnxt = tp;
    }

    const float bs = bias[e];
    float* outB = out + (size_t)b * SS * OO;
    #pragma unroll
    for (int mi = 0; mi < 4; ++mi) {
        #pragma unroll
        for (int v = 0; v < 8; ++v) {
            const int rl = wm * 64 + mi * 16 + v + 8 * hi;
            const int tok = sTok[rl];
            const float rc = sRec[rl];
            #pragma unroll
            for (int ni = 0; ni < 4; ++ni) {
                const int gc = nt * BN + wn * 64 + ni * 16 + l15;
                const float val = (acc[mi][ni][v] + bs) * rc;
                atomicAdd(outB + (size_t)tok * OO + gc, val);
            }
        }
    }
}

// ---------------------------------------------------------------------------
extern "C" void kernel_launch(void* const* d_in, const int* in_sizes, int n_in,
                              void* d_out, int out_size, void* d_ws, size_t ws_size,
                              hipStream_t stream) {
    (void)in_sizes; (void)n_in;
    const float* x    = (const float*)d_in[0];
    const float* gw   = (const float*)d_in[1];
    const float* gb   = (const float*)d_in[2];
    const float* w    = (const float*)d_in[3];
    const float* bias = (const float*)d_in[4];
    float* out = (float*)d_out;

    // workspace layout
    const size_t nLogits = (size_t)BB * EE * SS;     // 131072 f32
    const size_t nSel    = (size_t)BB * EE * KK;     // 32768
    float* logitsT = (float*)d_ws;
    float* probsT  = logitsT + nLogits;
    int*   idxW    = (int*)(probsT + nLogits);
    float* recipW  = (float*)(idxW + nSel);
    const size_t smallBytes = (2 * nLogits + 2 * nSel) * 4;     // 1,310,720 B
    const size_t xbElems = (size_t)BB * SS * DD;                // 8M bf16
    const size_t wTElems = (size_t)EE * DD * OO;                // 16M bf16
    unsigned short* xb = (unsigned short*)((char*)d_ws + smallBytes);
    unsigned short* wT = xb + xbElems;
    const size_t needBytes = smallBytes + (xbElems + wTElems) * 2;

    gate_logits<<<BB * SS, 128, 0, stream>>>(x, gw, gb, logitsT);
    softmax_tok<<<BB * EE, 256, 0, stream>>>(logitsT, probsT);
    topk_sel<<<BB * EE, 256, 0, stream>>>(probsT, idxW, recipW);
    hipMemsetAsync(d_out, 0, (size_t)out_size * sizeof(float), stream);

    if (ws_size >= needBytes) {
        cvt_x<<<(int)(xbElems / 4 / 256), 256, 0, stream>>>(x, xb);
        cvt_wT<<<(int)(wTElems / 8 / 256), 256, 0, stream>>>(w, wT);
        moe_gemm_bf<<<BB * EE * 16, 256, 0, stream>>>(xb, wT, bias, idxW, recipW, out);
    } else {
        moe_gemm<<<BB * EE * 16, 256, 0, stream>>>(x, w, bias, idxW, recipW, out);
    }
}
